// RSPPLoss_25915832664575
// MI455X (gfx1250) — matrix operand order repaired and checked by reading, not compiled, not run on hardware
//
#include <hip/hip_runtime.h>
#include <hip/hip_bf16.h>
#include <math.h>

#define N_ITEMS 65536
#define DIM     128
#define NCLS    2048

typedef __attribute__((ext_vector_type(16))) _Float16 v16h;
typedef __attribute__((ext_vector_type(8)))  float    v8f;

// ---------------------------------------------------------------- zero scratch
__global__ void rspp_zero_kernel(float* __restrict__ p, int n) {
    int i = blockIdx.x * blockDim.x + threadIdx.x;
    if (i < n) p[i] = 0.0f;
}

// ------------------------------------------------- segment sums via atomics
// one thread per (item, feature) element; 8M float atomics into 256K slots
__global__ void rspp_accum_kernel(const float* __restrict__ feat,
                                  const int*   __restrict__ mlab,
                                  const int*   __restrict__ slab,
                                  float* __restrict__ sums,
                                  float* __restrict__ counts) {
    int idx = blockIdx.x * blockDim.x + threadIdx.x;
    if (idx >= N_ITEMS * DIM) return;
    int i = idx >> 7;          // item
    int j = idx & (DIM - 1);   // feature
    if (slab[i] == 0) {
        int c = mlab[i];
        atomicAdd(&sums[(size_t)c * DIM + j], feat[idx]);
        if (j == 0) atomicAdd(&counts[c], 1.0f);
    }
}

// ------------------------------- mean -> L2 normalize -> f16 proto + present
// one wave32 per class, 8 classes per 256-thread block
__global__ void rspp_proto_kernel(const float* __restrict__ sums,
                                  const float* __restrict__ counts,
                                  _Float16* __restrict__ proto,
                                  float* __restrict__ present) {
    int wid  = threadIdx.x >> 5;
    int lane = threadIdx.x & 31;
    int c    = blockIdx.x * 8 + wid;
    if (c >= NCLS) return;

    float cnt     = counts[c];
    float inv_cnt = 1.0f / fmaxf(cnt, 1.0f);

    const float* s = sums + (size_t)c * DIM + lane * 4;
    float m[4];
    float ss = 0.0f;
#pragma unroll
    for (int t = 0; t < 4; ++t) { m[t] = s[t] * inv_cnt; ss += m[t] * m[t]; }
#pragma unroll
    for (int off = 1; off < 32; off <<= 1) ss += __shfl_xor(ss, off, 32);

    float inv_n = 1.0f / fmaxf(sqrtf(ss), 1e-12f);
    _Float16* p = proto + (size_t)c * DIM + lane * 4;
#pragma unroll
    for (int t = 0; t < 4; ++t) p[t] = (_Float16)(m[t] * inv_n);
    if (lane == 0) present[c] = (cnt > 0.0f) ? 1.0f : 0.0f;
}

// --------------------------- fused WMMA GEMM + masked-softmax ratio per row
// block = 16 rows (one M tile), 4 waves; each wave sweeps 32 of the 128
// column tiles with 4x v_wmma_f32_16x16x32_f16 per tile (K = 128).
// Numerator (own-label cell) and denominator are both kept in registers and
// reduced once at the end -- no predicated stores in the hot loop.
__global__ __launch_bounds__(128) void rspp_gemm_softmax_kernel(
        const float*    __restrict__ feat,
        const int*      __restrict__ mlab,
        const _Float16* __restrict__ proto,
        const float*    __restrict__ present,
        float*          __restrict__ out) {
    __shared__ float sPos[4][16];
    __shared__ float sDen[4][16];

    const int wave    = threadIdx.x >> 5;
    const int lane    = threadIdx.x & 31;
    const int hb      = lane >> 4;    // which 16-lane half
    const int l16     = lane & 15;
    const int rowbase = blockIdx.x << 4;

    // ---- A fragments: 16-bit A 16x32, contiguous k order (k = 16*half + s, same as B):
    // lanes 0-15: VGPR0-3 <- K 0-7,   VGPR4-7 <- K 8-15   (row M = lane)
    // lanes16-31: VGPR0-3 <- K 16-23, VGPR4-7 <- K 24-31  (row M = lane-16)
    const float* frow = feat + (size_t)(rowbase + l16) * DIM;
    v16h a[4];
#pragma unroll
    for (int kt = 0; kt < 4; ++kt) {
        const int kb = kt * 32 + hb * 16;
        v16h av;
#pragma unroll
        for (int h = 0; h < 8; ++h) {
            av[h]     = (_Float16)frow[kb + h];        // K = kb + h
            av[h + 8] = (_Float16)frow[kb + 8 + h];    // K = kb + 8 + h
        }
        a[kt] = av;
    }

    // labels of the 8 rows this lane's accumulators cover
    int lab[8];
#pragma unroll
    for (int v = 0; v < 8; ++v) lab[v] = mlab[rowbase + hb * 8 + v];

    float acc[8];   // denominator partials (this lane's columns)
    float pacc[8];  // numerator partials (nonzero in exactly one lane/tile)
#pragma unroll
    for (int v = 0; v < 8; ++v) { acc[v] = 0.0f; pacc[v] = 0.0f; }

    for (int t = wave; t < NCLS / 16; t += 4) {
        const int c = t * 16 + l16;                // this lane's class column
        const _Float16* prow = proto + (size_t)c * DIM;
        v8f d = {};
#pragma unroll
        for (int kt = 0; kt < 4; ++kt) {
            // B 32x16: lane's column = c, halves h=0..15 <- K = hb*16 + h
            v16h b = *(const v16h*)(prow + kt * 32 + hb * 16);
            d = __builtin_amdgcn_wmma_f32_16x16x32_f16(
                    false, a[kt], false, b, (short)0, d, false, false);
        }
        const float pres = present[c];
#pragma unroll
        for (int v = 0; v < 8; ++v) {
            // C/D layout: lane holds col c, rows M = hb*8 + v
            float e = pres * __expf(d[v]);
            acc[v]  += e;
            pacc[v] += (lab[v] == c) ? e : 0.0f;   // branch-free select+add
        }
    }

    // reduce across the 16 lanes of each half (xor offsets 1..8 stay in half)
#pragma unroll
    for (int v = 0; v < 8; ++v) {
#pragma unroll
        for (int off = 1; off < 16; off <<= 1) {
            acc[v]  += __shfl_xor(acc[v],  off, 32);
            pacc[v] += __shfl_xor(pacc[v], off, 32);
        }
    }
    if (l16 == 0) {
#pragma unroll
        for (int v = 0; v < 8; ++v) {
            sDen[wave][hb * 8 + v] = acc[v];
            sPos[wave][hb * 8 + v] = pacc[v];
        }
    }
    __syncthreads();

    if (threadIdx.x < 16) {
        int r = threadIdx.x;
        float den = sDen[0][r] + sDen[1][r] + sDen[2][r] + sDen[3][r];
        float pos = sPos[0][r] + sPos[1][r] + sPos[2][r] + sPos[3][r];
        int   l   = mlab[rowbase + r];
        out[rowbase + r] = (present[l] > 0.0f) ? (pos / den) : 1.0f;
    }
}

// ----------------------------------------------------------------- launcher
extern "C" void kernel_launch(void* const* d_in, const int* in_sizes, int n_in,
                              void* d_out, int out_size, void* d_ws, size_t ws_size,
                              hipStream_t stream) {
    const float* feat = (const float*)d_in[0];
    const int*   mlab = (const int*)d_in[1];
    const int*   slab = (const int*)d_in[2];
    float*       out  = (float*)d_out;

    // workspace layout
    char*  ws      = (char*)d_ws;
    float* sums    = (float*)ws;                               // C*D f32 (1 MB)
    float* counts  = (float*)(ws + (size_t)NCLS * DIM * 4);    // C f32
    float* present = counts + NCLS;                            // C f32
    _Float16* proto = (_Float16*)(present + NCLS);             // C*D f16 (32B-aligned)

    // 1) zero sums+counts (contiguous)
    {
        int n = NCLS * DIM + NCLS;
        rspp_zero_kernel<<<(n + 255) / 256, 256, 0, stream>>>(sums, n);
    }
    // 2) atomic segment sums
    {
        int n = N_ITEMS * DIM;
        rspp_accum_kernel<<<(n + 255) / 256, 256, 0, stream>>>(feat, mlab, slab,
                                                               sums, counts);
    }
    // 3) prototypes: mean + normalize -> f16, present flags
    rspp_proto_kernel<<<NCLS / 8, 256, 0, stream>>>(sums, counts, proto, present);

    // 4) fused WMMA GEMM + masked softmax ratio
    rspp_gemm_softmax_kernel<<<N_ITEMS / 16, 128, 0, stream>>>(feat, mlab, proto,
                                                               present, out);
}
